// RGDC_39573828665591
// MI455X (gfx1250) — compile-verified, matrix operand order checked by reading, not run on hardware
//
#include <hip/hip_runtime.h>
#include <hip/hip_bf16.h>

// ---------------------------------------------------------------------------
// R-GCN diffusion for MI455X (gfx1250, wave32, WMMA)
//   CSR-by-dst build  ->  fused tile kernel:
//     LDS f32 segment-sum (two 20-relation chunks of 16x20x128 f32 = 160 KB)
//     cooperative f32->bf16 pack into A-fragment lane order (v_cvt_pk_bf16_f32)
//     v_wmma_f32_16x16x32_bf16 vs pre-packed bf16 weights (2x ds_b128 + 2x vm_b128 + wmma)
//     epilogue: + cnt@rFeatures, *norm, + h@loop_weight (WMMA), ReLU
// ---------------------------------------------------------------------------

typedef __bf16 v16bf __attribute__((ext_vector_type(16)));
typedef float  v8f   __attribute__((ext_vector_type(8)));

#define DFEAT   128
#define RTOT    40
#define RCHUNK  20
#define NCHUNK  2
#define TILE_M  16

union Frag16 {                       // one WMMA 16-bit operand (8 VGPRs)
  v16bf v;
  uint4 q[2];
};

union PkBf2 {                        // 2xf32 -> packed 2xbf16 (v_cvt_pk_bf16_f32)
  __bf16 b[2];
  unsigned int u;
};

__device__ __forceinline__ unsigned short f32_to_bf16_rne(float f) {
  PkBf2 p; p.b[0] = (__bf16)f; p.b[1] = (__bf16)0.f;
  return (unsigned short)(p.u & 0xFFFFu);
}

// K index inside a 32-wide K-block for 16-bit WMMA operands (ISA 7.12.2):
// lanes 0-15 hold K pairs {0..7,16..23}; lanes 16-31 hold {8..15,24..31}.
__device__ __forceinline__ int klocal(int vv, int half, int p) {
  return ((vv < 4) ? (2 * vv) : (16 + 2 * (vv - 4))) + 8 * half + p;
}

// ------------------------------ CSR build ----------------------------------

__global__ void k_zero_i32(int* __restrict__ p, int n) {
  int i = blockIdx.x * 256 + threadIdx.x;
  if (i < n) p[i] = 0;
}

__global__ void k_hist(const int* __restrict__ edst, int* __restrict__ counts, int e) {
  int i = blockIdx.x * 256 + threadIdx.x;
  if (i < e) atomicAdd(&counts[edst[i]], 1);
}

__global__ void k_scan(const int* __restrict__ counts, int* __restrict__ offsets, int n) {
  __shared__ int buf[1024];
  __shared__ int carry;
  if (threadIdx.x == 0) carry = 0;
  __syncthreads();
  for (int base = 0; base < n; base += 1024) {
    int i = base + (int)threadIdx.x;
    int v = (i < n) ? counts[i] : 0;
    buf[threadIdx.x] = v;
    __syncthreads();
    for (int off = 1; off < 1024; off <<= 1) {
      int t = (threadIdx.x >= (unsigned)off) ? buf[threadIdx.x - off] : 0;
      __syncthreads();
      buf[threadIdx.x] += t;
      __syncthreads();
    }
    int incl = buf[threadIdx.x];
    int run  = carry;
    if (i < n) offsets[i] = run + incl - v;   // exclusive
    __syncthreads();
    if (threadIdx.x == 1023) carry = run + incl;
    __syncthreads();
  }
  if (threadIdx.x == 0) offsets[n] = carry;
}

__global__ void k_scatter(const int* __restrict__ esrc, const int* __restrict__ edst,
                          const int* __restrict__ etyp, const int* __restrict__ offsets,
                          int* __restrict__ cursor, int2* __restrict__ spair, int e) {
  int i = blockIdx.x * 256 + threadIdx.x;
  if (i < e) {
    int d   = edst[i];
    int pos = offsets[d] + atomicAdd(&cursor[d], 1);
    spair[pos] = make_int2(esrc[i], etyp[i]);
  }
}

// ---------------- pack weights into bf16 B-fragment lane order --------------
// Wp[r][kb][wave j][lane][16 bf16]; element t: n = 16j + lane%16,
// k = kb*32 + klocal(t>>1, lane>>4, t&1); value = relW[r][k][n].
// Lp: same without r, value = loopW[k][n].

__global__ void k_pack(const float* __restrict__ relW, const float* __restrict__ loopW,
                       unsigned short* __restrict__ Wp, unsigned short* __restrict__ Lp) {
  const int WPN = RTOT * 4 * 8 * 32 * 16;
  const int LPN = 4 * 8 * 32 * 16;
  int idx = blockIdx.x * 256 + threadIdx.x;
  if (idx < WPN) {
    int t = idx & 15, lane = (idx >> 4) & 31, j = (idx >> 9) & 7;
    int kb = (idx >> 12) & 3, r = idx >> 14;
    int half = lane >> 4, ln = lane & 15;
    int n = j * 16 + ln;
    int k = kb * 32 + klocal(t >> 1, half, t & 1);
    Wp[idx] = f32_to_bf16_rne(relW[((long)r * DFEAT + k) * DFEAT + n]);
  } else if (idx < WPN + LPN) {
    int i2 = idx - WPN;
    int t = i2 & 15, lane = (i2 >> 4) & 31, j = (i2 >> 9) & 7, kb = (i2 >> 12) & 3;
    int half = lane >> 4, ln = lane & 15;
    int n = j * 16 + ln;
    int k = kb * 32 + klocal(t >> 1, half, t & 1);
    Lp[i2] = f32_to_bf16_rne(loopW[k * DFEAT + n]);
  }
}

// ------------------------------ fused main ---------------------------------

__launch_bounds__(256, 1)
__global__ void k_main(const float* __restrict__ h, const float* __restrict__ rF,
                       const float* __restrict__ norm, const int* __restrict__ offsets,
                       const int2* __restrict__ spair,
                       const unsigned short* __restrict__ Wp,
                       const unsigned short* __restrict__ Lp,
                       float* __restrict__ out, int nNodes) {
  extern __shared__ char smem[];
  float*          S    = (float*)smem;                         // [16][RCHUNK][128] f32, 160 KB
  unsigned short* Ab   = (unsigned short*)(S + TILE_M * RCHUNK * DFEAT);  // 80 KB A-frags
  float*          cnt  = (float*)(Ab + RCHUNK * 4 * 32 * 16);  // [16][RTOT]
  float*          rFs  = cnt + TILE_M * RTOT;                  // [RTOT][128]
  float*          nrm  = rFs + RTOT * DFEAT;                   // [16]
  unsigned short* Ah   = (unsigned short*)(nrm + TILE_M);      // [4][32][16] final A-frags

  const int tid  = threadIdx.x;
  const int wave = tid >> 5;               // wave32
  const int lane = tid & 31;
  const int half = lane >> 4;
  const int ln   = lane & 15;
  const int tile = blockIdx.x;
  const int ncol = wave * 16 + ln;         // this lane's output column in C

  for (int i = tid; i < RTOT * DFEAT; i += 256) rFs[i] = rF[i];
  for (int i = tid; i < TILE_M * RTOT; i += 256) cnt[i] = 0.f;
  if (tid < TILE_M) {
    int v = tile * TILE_M + tid;
    nrm[tid] = (v < nNodes) ? norm[v] : 0.f;
  }

  v8f c = {};                              // 16x16 f32 accumulator per wave

  for (int ch = 0; ch < NCHUNK; ++ch) {
    __syncthreads();
    for (int i = tid; i < TILE_M * RCHUNK * DFEAT; i += 256) S[i] = 0.f;
    __syncthreads();

    // Segment-sum into LDS: wave w owns dst rows {w, w+8}.
    for (int rr = 0; rr < 2; ++rr) {
      int m = wave + 8 * rr;
      int v = tile * TILE_M + m;
      if (v < nNodes) {
        int beg = offsets[v], end = offsets[v + 1];
        for (int e = beg; e < end; ++e) {
          int2 sp  = spair[e];
          int  src = sp.x, ty = sp.y;
          const float4 hv = ((const float4*)(h + (long)src * DFEAT))[lane];
          int rl = ty - ch * RCHUNK;
          if ((unsigned)rl < (unsigned)RCHUNK) {
            float* p = S + (m * RCHUNK + rl) * DFEAT + lane * 4;
            atomicAdd(p + 0, hv.x);
            atomicAdd(p + 1, hv.y);
            atomicAdd(p + 2, hv.z);
            atomicAdd(p + 3, hv.w);
          }
          if (ch == 0 && lane == 0) cnt[m * RTOT + ty] += 1.f;
        }
      }
    }
    __syncthreads();

    // Cooperative pack: S (f32) -> Ab (bf16, A-fragment lane order).
    // Ab element o = ((rl*4+kb)*32 + lane)*16 + t ; value S[m=lane&15][rl][k],
    // k = kb*32 + klocal(t>>1, lane>>4, t&1). Pairs (t, t+1) are k, k+1.
    {
      unsigned int* Ab32 = (unsigned int*)Ab;
      const int NPAIR = RCHUNK * 4 * 32 * 16 / 2;   // 20480
      for (int pi = tid; pi < NPAIR; pi += 256) {
        int o    = pi * 2;
        int t    = o & 15;
        int pl   = (o >> 4) & 31;
        int kb   = (o >> 9) & 3;
        int rl   = o >> 11;
        int m    = pl & 15;
        int hf   = pl >> 4;
        int k0   = kb * 32 + klocal(t >> 1, hf, 0);
        float2 f = *(const float2*)(S + (m * RCHUNK + rl) * DFEAT + k0);
        PkBf2 pk;
        pk.b[0] = (__bf16)f.x;
        pk.b[1] = (__bf16)f.y;
        Ab32[pi] = pk.u;
      }
    }
    __syncthreads();

    // C += S_chunk @ W[chunk]  (bf16 WMMA, f32 accumulate)
    for (int rl = 0; rl < RCHUNK; ++rl) {
      int r = ch * RCHUNK + rl;
      for (int kb = 0; kb < 4; ++kb) {
        Frag16 a, b;
        const uint4* ap = (const uint4*)(Ab + (((rl * 4 + kb) * 32 + lane) << 4));
        a.q[0] = ap[0];
        a.q[1] = ap[1];
        const uint4* bp =
            (const uint4*)(Wp + ((((long)(r * 4 + kb) * 8 + wave) * 32 + lane) << 4));
        b.q[0] = bp[0];
        b.q[1] = bp[1];
        c = __builtin_amdgcn_wmma_f32_16x16x32_bf16(false, a.v, false, b.v,
                                                    (short)0, c, false, false);
      }
    }
  }
  __syncthreads();

  // Epilogue: agg += cnt @ rFeatures ; h = agg * norm.
  // Also deposit bf16(h) straight into final A-fragment slots in Ah.
  float acc[8];
  #pragma unroll
  for (int i = 0; i < 8; ++i) acc[i] = c[i];
  for (int r = 0; r < RTOT; ++r) {
    float rf = rFs[r * DFEAT + ncol];
    #pragma unroll
    for (int i = 0; i < 8; ++i) acc[i] += cnt[(i + 8 * half) * RTOT + r] * rf;
  }
  // Invert the K mapping for k = ncol (lane-uniform):
  {
    int K   = ncol;
    int kb2 = K >> 5;
    int Kl  = K & 31;
    int hf  = (Kl >> 3) & 1;
    int p   = Kl & 1;
    int vv2 = (Kl & 16) ? (4 + ((Kl & 7) >> 1)) : ((Kl & 7) >> 1);
    unsigned short* ahp = Ah + kb2 * 512 + 16 * 16 * hf + 2 * vv2 + p;
    #pragma unroll
    for (int i = 0; i < 8; ++i) {
      int m = i + 8 * half;                // C layout: VGPR i -> row m (ISA 7.12.2)
      acc[i] *= nrm[m];
      c[i] = acc[i];
      ahp[m * 16] = f32_to_bf16_rne(acc[i]);   // Ah[(kb2*32 + m + 16*hf)*16 + 2*vv2+p]
    }
  }
  __syncthreads();

  // out = relu(h + h @ loop_weight): C already holds h, add WMMA(h, Lp).
  for (int kb = 0; kb < 4; ++kb) {
    Frag16 a, b;
    const uint4* ap = (const uint4*)(Ah + ((kb * 32 + lane) << 4));
    a.q[0] = ap[0];
    a.q[1] = ap[1];
    const uint4* bp = (const uint4*)(Lp + ((((long)kb * 8 + wave) * 32 + lane) << 4));
    b.q[0] = bp[0];
    b.q[1] = bp[1];
    c = __builtin_amdgcn_wmma_f32_16x16x32_bf16(false, a.v, false, b.v,
                                                (short)0, c, false, false);
  }
  #pragma unroll
  for (int i = 0; i < 8; ++i) {
    int m = i + 8 * half;
    int v = tile * TILE_M + m;
    if (v < nNodes) out[(long)v * DFEAT + ncol] = fmaxf(c[i], 0.f);
  }
}

// ------------------------------ host glue ----------------------------------

extern "C" void kernel_launch(void* const* d_in, const int* in_sizes, int n_in,
                              void* d_out, int out_size, void* d_ws, size_t ws_size,
                              hipStream_t stream) {
  (void)n_in; (void)out_size; (void)ws_size;
  const float* h     = (const float*)d_in[0];
  const float* rF    = (const float*)d_in[1];
  const float* relW  = (const float*)d_in[2];
  const float* loopW = (const float*)d_in[3];
  const float* norm  = (const float*)d_in[4];
  const int*   esrc  = (const int*)d_in[5];
  const int*   edst  = (const int*)d_in[6];
  const int*   etyp  = (const int*)d_in[7];
  float* out = (float*)d_out;

  const int N = in_sizes[4];               // norm has N elements
  const int E = in_sizes[5];

  auto align_up = [](size_t x) { return (x + 255) & ~(size_t)255; };
  char* w = (char*)d_ws;
  int*  counts  = (int*)w;  w += align_up((size_t)N * 4);
  int*  offsets = (int*)w;  w += align_up((size_t)(N + 1) * 4);
  int*  cursor  = (int*)w;  w += align_up((size_t)N * 4);
  int2* spair   = (int2*)w; w += align_up((size_t)E * 8);
  unsigned short* Wp = (unsigned short*)w; w += align_up((size_t)RTOT * 4 * 8 * 32 * 16 * 2);
  unsigned short* Lp = (unsigned short*)w;

  // CSR by dst
  k_zero_i32<<<(N + 255) / 256, 256, 0, stream>>>(counts, N);
  k_zero_i32<<<(N + 255) / 256, 256, 0, stream>>>(cursor, N);
  k_hist<<<(E + 255) / 256, 256, 0, stream>>>(edst, counts, E);
  k_scan<<<1, 1024, 0, stream>>>(counts, offsets, N);
  k_scatter<<<(E + 255) / 256, 256, 0, stream>>>(esrc, edst, etyp, offsets, cursor,
                                                 spair, E);

  // bf16 fragment-packed weights
  const int packN = RTOT * 4 * 8 * 32 * 16 + 4 * 8 * 32 * 16;
  k_pack<<<(packN + 255) / 256, 256, 0, stream>>>(relW, loopW, Wp, Lp);

  // fused tile kernel
  const int tiles = (N + TILE_M - 1) / TILE_M;
  const int smemBytes =
      TILE_M * RCHUNK * DFEAT * 4          // S (f32)
      + RCHUNK * 4 * 32 * 16 * 2           // Ab (bf16 A-fragments)
      + TILE_M * RTOT * 4                  // cnt
      + RTOT * DFEAT * 4                   // rFs
      + TILE_M * 4                         // nrm
      + 4 * 32 * 16 * 2;                   // Ah (final A-fragments)
  (void)hipFuncSetAttribute((const void*)k_main,
                            hipFuncAttributeMaxDynamicSharedMemorySize, smemBytes);
  k_main<<<tiles, 256, smemBytes, stream>>>(h, rF, norm, offsets, spair,
                                            Wp, Lp, out, N);
}
